// MultiHeadAttention_4810363372421
// MI455X (gfx1250) — matrix operand order repaired and checked by reading, not compiled, not run on hardware
//
#include <hip/hip_runtime.h>
#include <math.h>

// Problem constants (fixed by the reference harness)
#define B_   2
#define L_   2048
#define D_   1024
#define H_   16
#define DH_  64
#define BL_  (B_ * L_)      // 4096 rows
#define QKV_ (3 * D_)       // 3072 cols

typedef __attribute__((ext_vector_type(16))) _Float16 v16h;
typedef __attribute__((ext_vector_type(8)))  _Float16 v8h;
typedef __attribute__((ext_vector_type(4)))  _Float16 v4h;
typedef __attribute__((ext_vector_type(8)))  float    v8f;
typedef __attribute__((ext_vector_type(4)))  float    v4f;

// ---------------------------------------------------------------------------
// Fragment loaders for V_WMMA_F32_16X16X32_F16 (wave32).
//
// A-matrix 16x32 f16 (ISA 7.12.2): lane l<16 holds row (rowBase+l),
// K = {kBase+0..7, kBase+16..23}; lane l>=16 holds the same row index with
// K = {kBase+8..15, kBase+24..31}. Two contiguous 16B LDS reads per lane.
__device__ __forceinline__ v16h load_frag_a(const _Float16* tile, int stride,
                                            int rowBase, int kBase) {
  const int lane = threadIdx.x & 31;
  const int lr = lane & 15, hf = lane >> 4;
  const _Float16* p = tile + (rowBase + lr) * stride + kBase + hf * 8;
  v8h lo = *(const v8h*)(p);
  v8h hi = *(const v8h*)(p + 16);
  v16h r;
#pragma unroll
  for (int i = 0; i < 8; ++i) { r[i] = lo[i]; r[8 + i] = hi[i]; }
  return r;
}

// B-matrix 32x16 f16: lane l<16 holds column (colBase+l), K = {kBase+0..7,
// kBase+16..23}; lane l>=16 holds K = {kBase+8..15, kBase+24..31} (per the sparse-B tables,
// halved for dense K=32). Read from a transposed LDS tile tileT[n][k] so the
// 32 bytes per lane are two contiguous 16B runs.
__device__ __forceinline__ v16h load_frag_b(const _Float16* tileT, int stride,
                                            int colBase, int kBase) {
  const int lane = threadIdx.x & 31;
  const int lr = lane & 15, hf = lane >> 4;
  const _Float16* p = tileT + (colBase + lr) * stride + kBase + hf * 8;
  v8h lo = *(const v8h*)(p);
  v8h hi = *(const v8h*)(p + 16);
  v16h r;
#pragma unroll
  for (int i = 0; i < 8; ++i) { r[i] = lo[i]; r[8 + i] = hi[i]; }
  return r;
}

// ---------------------------------------------------------------------------
// Generic C = A * B + bias GEMM, f32 in / f32 out, f16 WMMA compute.
// Block tile 128(M) x 128(N) x 32(K); 8 waves, each wave owns 32x64 of C
// (2x4 grid of 16x16 accumulators => 8 v_wmma per K-step per wave).
// Double-buffered LDS + register-prefetch pipeline: global loads for tile
// k+1 are in flight while tile k is consumed by WMMAs; one barrier per step.
#define GLDS 40  // padded LDS row stride (f16 elems); 80 B = 16B multiple

__global__ __launch_bounds__(256) void wmma_gemm_f32f16(
    const float* __restrict__ A, const float* __restrict__ Bm,
    const float* __restrict__ bias, float* __restrict__ C,
    int M, int N, int K, int lda, int ldb, int ldc, int ntStore) {
  __shared__ __align__(16) _Float16 As[2][128 * GLDS];   // A tile, row-major
  __shared__ __align__(16) _Float16 BsT[2][128 * GLDS];  // B tile, [n][k]

  const int t = threadIdx.x;
  const int wave = t >> 5;
  const int wm = wave >> 1, wn = wave & 1;
  const int m0 = blockIdx.y * 128;
  const int n0 = blockIdx.x * 128;

  // Per-thread staging coordinates.
  const int ac4 = (t & 7) * 4, ar = t >> 3;   // A: rows ar+{0,32,64,96}
  const int bc4 = (t & 31) * 4, br = t >> 5;  // B: rows br+{0,8,16,24}

  v4f ra[4], rb[4];

  // Issue all global loads for one K-tile back-to-back (8 outstanding).
  auto issue_loads = [&](int k0) {
#pragma unroll
    for (int j = 0; j < 4; ++j)
      ra[j] = *(const v4f*)(A + (size_t)(m0 + ar + 32 * j) * lda + k0 + ac4);
#pragma unroll
    for (int j = 0; j < 4; ++j)
      rb[j] = *(const v4f*)(Bm + (size_t)(k0 + br + 8 * j) * ldb + n0 + bc4);
  };
  // Convert f32 -> f16 and commit to LDS buffer `buf`.
  auto commit_lds = [&](int buf) {
#pragma unroll
    for (int j = 0; j < 4; ++j) {
      v4h h;
      h[0] = (_Float16)ra[j][0]; h[1] = (_Float16)ra[j][1];
      h[2] = (_Float16)ra[j][2]; h[3] = (_Float16)ra[j][3];
      *(v4h*)(&As[buf][(ar + 32 * j) * GLDS + ac4]) = h;
    }
#pragma unroll
    for (int j = 0; j < 4; ++j) {
      BsT[buf][(bc4 + 0) * GLDS + br + 8 * j] = (_Float16)rb[j][0];
      BsT[buf][(bc4 + 1) * GLDS + br + 8 * j] = (_Float16)rb[j][1];
      BsT[buf][(bc4 + 2) * GLDS + br + 8 * j] = (_Float16)rb[j][2];
      BsT[buf][(bc4 + 3) * GLDS + br + 8 * j] = (_Float16)rb[j][3];
    }
  };

  v8f acc[2][4] = {};

  issue_loads(0);
  commit_lds(0);
  __syncthreads();

  const int nk = K >> 5;
  for (int ki = 0; ki < nk; ++ki) {
    const int buf = ki & 1;
    if (ki + 1 < nk) issue_loads((ki + 1) << 5);  // in flight during compute

    v16h af[2];
#pragma unroll
    for (int ms = 0; ms < 2; ++ms)
      af[ms] = load_frag_a(As[buf], GLDS, wm * 32 + ms * 16, 0);
#pragma unroll
    for (int ns = 0; ns < 4; ++ns) {
      v16h bf = load_frag_b(BsT[buf], GLDS, wn * 64 + ns * 16, 0);
#pragma unroll
      for (int ms = 0; ms < 2; ++ms)
        acc[ms][ns] = __builtin_amdgcn_wmma_f32_16x16x32_f16(
            false, af[ms], false, bf, (short)0, acc[ms][ns], false, false);
    }

    if (ki + 1 < nk) commit_lds(buf ^ 1);  // waits the loads here
    __syncthreads();
  }

  // Epilogue: C/D layout -> VGPR i holds row i (lanes 0-15) / i+8 (16-31).
  const int lane = t & 31;
  const int lr = lane & 15, hf = lane >> 4;
#pragma unroll
  for (int ms = 0; ms < 2; ++ms)
#pragma unroll
    for (int ns = 0; ns < 4; ++ns) {
      const int gc = n0 + wn * 64 + ns * 16 + lr;
      const float bv = bias[gc];
#pragma unroll
      for (int i = 0; i < 8; ++i) {
        const int gr = m0 + wm * 32 + ms * 16 + hf * 8 + i;
        float* p = C + (size_t)gr * ldc + gc;
        const float v = acc[ms][ns][i] + bv;
        if (ntStore) __builtin_nontemporal_store(v, p);
        else *p = v;
      }
    }
}

// ---------------------------------------------------------------------------
// Flash attention, causal, per (b, h). Block = 128 threads (4 waves), each
// block handles 64 query rows; wave w owns 16 of them. Br = Bc = 64.
// S = (Q K^T)/8 with online softmax; P routed through LDS (C-layout ->
// A-layout reshape) for the P*V WMMA.
#define ALDS 72  // padded LDS row stride (f16); 144 B = 16B multiple

__global__ __launch_bounds__(128) void flash_attn_wmma(
    const float* __restrict__ qkv, float* __restrict__ attnOut) {
  __shared__ __align__(16) _Float16 Qs[64 * ALDS];       // [qrow][dh]
  __shared__ __align__(16) _Float16 Ks[64 * ALDS];       // [kvrow][dh]
  __shared__ __align__(16) _Float16 Vt[64 * ALDS];       // [dh][kvrow]
  __shared__ __align__(16) _Float16 Ps[4 * 16 * ALDS];   // per-wave P strip

  const int t = threadIdx.x;
  const int wave = t >> 5;
  const int lane = t & 31;
  const int lr = lane & 15, hf = lane >> 4;

  const int qt = blockIdx.x;            // query tile (64 rows)
  const int bh = blockIdx.y;
  const int b = bh >> 4, h = bh & 15;
  const int rowBase = b * L_;
  const int q0 = qt * 64;
  const int qcol = h * DH_;
  const int kcol = D_ + h * DH_;
  const int vcol = 2 * D_ + h * DH_;

  const int c4 = (t & 15) * 4;
  const int r0 = t >> 4;  // 0..7

  // Stage Q tile (f32 -> f16), batched loads first.
  {
    v4f rq[8];
#pragma unroll
    for (int j = 0; j < 8; ++j)
      rq[j] = *(const v4f*)(
          qkv + (size_t)(rowBase + q0 + r0 + 8 * j) * QKV_ + qcol + c4);
#pragma unroll
    for (int j = 0; j < 8; ++j) {
      v4h hv;
      hv[0] = (_Float16)rq[j][0]; hv[1] = (_Float16)rq[j][1];
      hv[2] = (_Float16)rq[j][2]; hv[3] = (_Float16)rq[j][3];
      *(v4h*)(&Qs[(r0 + 8 * j) * ALDS + c4]) = hv;
    }
  }
  __syncthreads();

  // Q A-fragments stay resident for the whole kernel.
  v16h aq[2];
  aq[0] = load_frag_a(Qs, ALDS, wave * 16, 0);
  aq[1] = load_frag_a(Qs, ALDS, wave * 16, 32);

  float m[8], l[8];
  v8f o[4] = {};
#pragma unroll
  for (int i = 0; i < 8; ++i) { m[i] = -INFINITY; l[i] = 0.0f; }

  for (int ct = 0; ct <= qt; ++ct) {   // causal: skip fully-masked tiles
    const int c0 = ct * 64;
    // Stage K tile row-major and V tile transposed; 8 loads in flight each.
    {
      v4f rk[8];
#pragma unroll
      for (int j = 0; j < 8; ++j)
        rk[j] = *(const v4f*)(
            qkv + (size_t)(rowBase + c0 + r0 + 8 * j) * QKV_ + kcol + c4);
      v4f rv[8];
#pragma unroll
      for (int j = 0; j < 8; ++j)
        rv[j] = *(const v4f*)(
            qkv + (size_t)(rowBase + c0 + r0 + 8 * j) * QKV_ + vcol + c4);
#pragma unroll
      for (int j = 0; j < 8; ++j) {
        v4h hk;
        hk[0] = (_Float16)rk[j][0]; hk[1] = (_Float16)rk[j][1];
        hk[2] = (_Float16)rk[j][2]; hk[3] = (_Float16)rk[j][3];
        *(v4h*)(&Ks[(r0 + 8 * j) * ALDS + c4]) = hk;
      }
#pragma unroll
      for (int j = 0; j < 8; ++j) {
        const int r = r0 + 8 * j;
        Vt[(c4 + 0) * ALDS + r] = (_Float16)rv[j][0];
        Vt[(c4 + 1) * ALDS + r] = (_Float16)rv[j][1];
        Vt[(c4 + 2) * ALDS + r] = (_Float16)rv[j][2];
        Vt[(c4 + 3) * ALDS + r] = (_Float16)rv[j][3];
      }
    }
    __syncthreads();

    // S = Q * K^T. B = K^T indexed [n=kvrow][k=dh] == Ks row-major.
    v8f s[4] = {};
#pragma unroll
    for (int ns = 0; ns < 4; ++ns) {
      v16h bk0 = load_frag_b(Ks, ALDS, ns * 16, 0);
      s[ns] = __builtin_amdgcn_wmma_f32_16x16x32_f16(
          false, aq[0], false, bk0, (short)0, s[ns], false, false);
      v16h bk1 = load_frag_b(Ks, ALDS, ns * 16, 32);
      s[ns] = __builtin_amdgcn_wmma_f32_16x16x32_f16(
          false, aq[1], false, bk1, (short)0, s[ns], false, false);
    }

    // Scale + causal mask (only the diagonal tile straddles the boundary).
    const bool diag = (ct == qt);
#pragma unroll
    for (int ns = 0; ns < 4; ++ns)
#pragma unroll
      for (int i = 0; i < 8; ++i) {
        float v = s[ns][i] * 0.125f;  // 1/sqrt(64)
        if (diag) {
          const int row = q0 + wave * 16 + hf * 8 + i;
          const int col = c0 + ns * 16 + lr;
          if (col > row) v = -INFINITY;
        }
        s[ns][i] = v;
      }

    // Online softmax. Row values live in a 16-lane group; xor masks 1,2,4,8
    // reduce within that group under wave32.
#pragma unroll
    for (int i = 0; i < 8; ++i) {
      float mx = fmaxf(fmaxf(s[0][i], s[1][i]), fmaxf(s[2][i], s[3][i]));
#pragma unroll
      for (int off = 8; off >= 1; off >>= 1)
        mx = fmaxf(mx, __shfl_xor(mx, off));
      const float mn = fmaxf(m[i], mx);
      const float corr = __expf(m[i] - mn);
      float ps = 0.0f;
#pragma unroll
      for (int ns = 0; ns < 4; ++ns) {
        const float p = __expf(s[ns][i] - mn);
        Ps[(wave * 16 + hf * 8 + i) * ALDS + ns * 16 + lr] = (_Float16)p;
        ps += p;
      }
#pragma unroll
      for (int off = 8; off >= 1; off >>= 1)
        ps += __shfl_xor(ps, off);
      l[i] = l[i] * corr + ps;
      m[i] = mn;
#pragma unroll
      for (int ns = 0; ns < 4; ++ns) o[ns][i] *= corr;
    }

    // O += P * V. A = P strip from LDS; B = V indexed [n=dh][k=kvrow] == Vt.
#pragma unroll
    for (int kc = 0; kc < 2; ++kc) {
      v16h ap = load_frag_a(&Ps[wave * 16 * ALDS], ALDS, 0, kc * 32);
#pragma unroll
      for (int ns = 0; ns < 4; ++ns) {
        v16h bv = load_frag_b(Vt, ALDS, ns * 16, kc * 32);
        o[ns] = __builtin_amdgcn_wmma_f32_16x16x32_f16(
            false, ap, false, bv, (short)0, o[ns], false, false);
      }
    }
    __syncthreads();  // all waves done with Ks/Vt before restaging
  }

  // Normalize and write [B, L, H, Dh] == [BL_, D_].
#pragma unroll
  for (int ns = 0; ns < 4; ++ns)
#pragma unroll
    for (int i = 0; i < 8; ++i) {
      const int row = rowBase + q0 + wave * 16 + hf * 8 + i;
      const int col = h * DH_ + ns * 16 + lr;
      attnOut[(size_t)row * D_ + col] = o[ns][i] / l[i];
    }
}

// ---------------------------------------------------------------------------
// Split the k/v slices of qkv into the tuple outputs (pure bandwidth).
// Outputs are never re-read -> non-temporal stores keep L2 for qkv.
__global__ __launch_bounds__(256) void copy_kv(const float* __restrict__ qkv,
                                               float* __restrict__ kOut,
                                               float* __restrict__ vOut) {
  const int idx = blockIdx.x * 256 + threadIdx.x;  // over BL_ * (D_/4)
  const int row = idx >> 8;
  const int c4 = (idx & 255) * 4;
  const size_t src = (size_t)row * QKV_;
  const size_t dst = (size_t)row * D_ + c4;
  const v4f kv = *(const v4f*)(qkv + src + D_ + c4);
  const v4f vv = *(const v4f*)(qkv + src + 2 * D_ + c4);
  __builtin_nontemporal_store(kv, (v4f*)(kOut + dst));
  __builtin_nontemporal_store(vv, (v4f*)(vOut + dst));
}

// ---------------------------------------------------------------------------
extern "C" void kernel_launch(void* const* d_in, const int* in_sizes, int n_in,
                              void* d_out, int out_size, void* d_ws,
                              size_t ws_size, hipStream_t stream) {
  (void)in_sizes; (void)n_in; (void)out_size; (void)ws_size;
  const float* inp  = (const float*)d_in[0];
  // d_in[1] is the causal mask; causality is hardcoded in flash_attn_wmma.
  const float* Wqkv = (const float*)d_in[2];
  const float* bqkv = (const float*)d_in[3];
  const float* Wout = (const float*)d_in[4];
  const float* bout = (const float*)d_in[5];

  float* out  = (float*)d_out;                     // [BL_, D_]
  float* kOut = out + (size_t)BL_ * D_;            // [BL_, D_]
  float* vOut = kOut + (size_t)BL_ * D_;           // [BL_, D_]

  float* qkv  = (float*)d_ws;                      // [BL_, 3072] f32 (48 MB)
  float* attn = qkv + (size_t)BL_ * QKV_;          // [BL_, 1024] f32 (16 MB)

  // 1) qkv = input @ W_qkv + b_qkv (re-read by attention -> regular stores)
  wmma_gemm_f32f16<<<dim3(QKV_ / 128, BL_ / 128), 256, 0, stream>>>(
      inp, Wqkv, bqkv, qkv, BL_, QKV_, D_, D_, QKV_, QKV_, /*nt=*/0);
  // 2) tuple outputs k, v (slices of qkv)
  copy_kv<<<BL_ * (D_ / 4) / 256, 256, 0, stream>>>(qkv, kOut, vOut);
  // 3) causal flash attention per (b, h)
  flash_attn_wmma<<<dim3(L_ / 64, B_ * H_), 128, 0, stream>>>(qkv, attn);
  // 4) out = attn @ W_out + b_out (write-only -> non-temporal stores)
  wmma_gemm_f32f16<<<dim3(D_ / 128, BL_ / 128), 256, 0, stream>>>(
      attn, Wout, bout, out, BL_, D_, D_, D_, D_, D_, /*nt=*/1);
}
